// AttentionBlock3D_14654428414489
// MI455X (gfx1250) — matrix operand order repaired and checked by reading, not compiled, not run on hardware
//
#include <hip/hip_runtime.h>
#include <math.h>

typedef _Float16 h8   __attribute__((ext_vector_type(8)));
typedef _Float16 v16h __attribute__((ext_vector_type(16)));
typedef float    v8f  __attribute__((ext_vector_type(8)));

#define HEADS    8
#define DHEAD    32
#define C_DIM    256
#define N_DIM    4096
#define O_QKV    768
#define EPS_GN   1e-5f
#define QSCALE   0.17677669529663687f   /* 32^-0.5 folded into W_q */

__device__ __forceinline__ v16h cat8(h8 lo, h8 hi) {
  return __builtin_shufflevector(lo, hi, 0,1,2,3,4,5,6,7,8,9,10,11,12,13,14,15);
}

#define WMMA_F16(A, B, C) \
  __builtin_amdgcn_wmma_f32_16x16x32_f16(false, (A), false, (B), (short)0, (C), false, false)

/* ---------------- GroupNorm stats -> per-channel scale/bias ------------- */
__global__ __launch_bounds__(256) void gn_stats(const float* __restrict__ x,
                                                const float* __restrict__ gw,
                                                const float* __restrict__ gb,
                                                float* __restrict__ sbuf,
                                                float* __restrict__ tbuf) {
  __shared__ float red0[256];
  __shared__ float red1[256];
  const int g = blockIdx.x;                     // 8 groups, 32 ch each
  const float* xg = x + (size_t)g * 32 * N_DIM; // channels contiguous per group
  float s = 0.f, ss = 0.f;
  for (int i = threadIdx.x; i < 32 * N_DIM; i += 256) {
    float v = xg[i];
    s += v; ss += v * v;
  }
  red0[threadIdx.x] = s; red1[threadIdx.x] = ss;
  __syncthreads();
  for (int st = 128; st > 0; st >>= 1) {
    if (threadIdx.x < st) {
      red0[threadIdx.x] += red0[threadIdx.x + st];
      red1[threadIdx.x] += red1[threadIdx.x + st];
    }
    __syncthreads();
  }
  if (threadIdx.x < 32) {
    const float inv_n = 1.0f / (32.0f * N_DIM);
    float mean = red0[0] * inv_n;
    float var  = red1[0] * inv_n - mean * mean;
    float rstd = rsqrtf(var + EPS_GN);
    int c = g * 32 + threadIdx.x;
    float sc = gw[c] * rstd;
    sbuf[c] = sc;
    tbuf[c] = gb[c] - mean * sc;
  }
}

/* ---------------- f32 -> f16 conversions -------------------------------- */
__global__ __launch_bounds__(256) void cvt_wqkv(const float* __restrict__ w,
                                                _Float16* __restrict__ w16) {
  int i = blockIdx.x * 256 + threadIdx.x;       // 768*256
  float sc = (i < 256 * C_DIM) ? QSCALE : 1.0f; // fold softmax scale into q rows
  w16[i] = (_Float16)(w[i] * sc);
}

__global__ __launch_bounds__(256) void cvt_wout(const float* __restrict__ w,
                                                _Float16* __restrict__ w16) {
  int i = blockIdx.x * 256 + threadIdx.x;       // 256*256
  w16[i] = (_Float16)w[i];
}

/* normalized x, stored transposed: xnT[n][c], f16 */
__global__ __launch_bounds__(256) void cvt_x(const float* __restrict__ x,
                                             const float* __restrict__ sbuf,
                                             const float* __restrict__ tbuf,
                                             _Float16* __restrict__ xnT) {
  int i = blockIdx.x * 256 + threadIdx.x;       // C*N
  int c = i >> 12;
  int n = i & (N_DIM - 1);
  xnT[(size_t)n * C_DIM + c] = (_Float16)(x[i] * sbuf[c] + tbuf[c]);
}

/* ---------------- QKV GEMM: qkvT[n][o] = (W_qkv @ xn)^T ----------------- */
/* 2x2 register blocking: each wave computes a 32(o) x 32(n) block.        */
__global__ __launch_bounds__(256) void gemm_qkv(const _Float16* __restrict__ w16,
                                                const _Float16* __restrict__ xnT,
                                                _Float16* __restrict__ qkvT,
                                                _Float16* __restrict__ vN) {
  const int lane = threadIdx.x;
  const int omac = blockIdx.y * 8 + threadIdx.y;    // 24 o-macros of 32 rows
  const int nmac = blockIdx.x;                      // 128 n-macros of 32 cols
  const int hf  = lane >> 4;
  const int l16 = lane & 15;
  const int arow0 = omac * 32 + l16, arow1 = arow0 + 16;
  const int bcol0 = nmac * 32 + l16, bcol1 = bcol0 + 16;
  v8f acc00 = {}, acc01 = {}, acc10 = {}, acc11 = {};
#pragma unroll
  for (int kk = 0; kk < C_DIM; kk += 32) {
    const h8* ap0 = (const h8*)(w16 + (size_t)arow0 * C_DIM + kk + hf * 16);
    const h8* ap1 = (const h8*)(w16 + (size_t)arow1 * C_DIM + kk + hf * 16);
    v16h a0 = cat8(ap0[0], ap0[1]);
    v16h a1 = cat8(ap1[0], ap1[1]);
    v16h b0 = *(const v16h*)(xnT + (size_t)bcol0 * C_DIM + kk + hf * 16);
    v16h b1 = *(const v16h*)(xnT + (size_t)bcol1 * C_DIM + kk + hf * 16);
    acc00 = WMMA_F16(a0, b0, acc00);
    acc01 = WMMA_F16(a0, b1, acc01);
    acc10 = WMMA_F16(a1, b0, acc10);
    acc11 = WMMA_F16(a1, b1, acc11);
  }
  const int ob0 = omac * 32 + hf * 8, ob1 = ob0 + 16;
  const int n0 = nmac * 32 + l16, n1 = n0 + 16;
  v8f accs[4] = {acc00, acc01, acc10, acc11};
  const int obs[4] = {ob0, ob0, ob1, ob1};
  const int ns[4]  = {n0, n1, n0, n1};
#pragma unroll
  for (int t = 0; t < 4; ++t) {
    h8 r;
#pragma unroll
    for (int i = 0; i < 8; ++i) r[i] = (_Float16)accs[t][i];
    *(h8*)(qkvT + (size_t)ns[t] * O_QKV + obs[t]) = r;   /* packed 16B store */
    if (obs[t] >= 512) {                                  /* duplicate V as [d][n] */
      int d = obs[t] - 512;
#pragma unroll
      for (int i = 0; i < 8; ++i) vN[(size_t)(d + i) * N_DIM + ns[t]] = r[i];
    }
  }
}

/* ---------------- Flash attention -------------------------------------- */
/* Block = 8 waves covering 128 i-rows of one head; K/V chunks (64 j) are  */
/* cooperatively staged into LDS and shared by all 8 waves (8x less L2).   */
__global__ __launch_bounds__(256) void flash_attn(const _Float16* __restrict__ qkvT,
                                                  const _Float16* __restrict__ vN,
                                                  _Float16* __restrict__ outT) {
  __shared__ __attribute__((aligned(32))) _Float16 ldsK[64 * 32];      /* kT[j][d]  */
  __shared__ __attribute__((aligned(32))) _Float16 ldsV[32 * 64];      /* v[d][j]   */
  __shared__ __attribute__((aligned(32))) _Float16 ldsP[8 * 16 * 64];  /* per-wave P */
  const int lane = threadIdx.x;
  const int w    = threadIdx.y;
  const int flat = w * 32 + lane;
  const int h    = blockIdx.y;
  const int i0   = blockIdx.x * 128 + w * 16;
  const int hf   = lane >> 4;
  const int l16  = lane & 15;

  /* Q as WMMA A operand (M=i, K=d=32), loaded once per wave */
  const _Float16* qb = qkvT + (size_t)(i0 + l16) * O_QKV + h * DHEAD + hf * 16;
  v16h a_q = cat8(*(const h8*)qb, *(const h8*)(qb + 8));

  float m_run[8], l_run[8];
  v8f o0 = {}, o1 = {};
#pragma unroll
  for (int r = 0; r < 8; ++r) { m_run[r] = -3.0e38f; l_run[r] = 0.f; }

  /* staging addresses (per flat thread, one h8 each) */
  const int jrK = flat >> 2, c8K = (flat & 3) * 8;          /* K: 64 x 32  */
  const int dV  = flat >> 3, j8V = (flat & 7) * 8;          /* V: 32 x 64  */
  const _Float16* kSrc = qkvT + 256 + h * DHEAD + c8K;
  const _Float16* vSrc = vN + (size_t)(h * DHEAD + dV) * N_DIM + j8V;

  for (int j0 = 0; j0 < N_DIM; j0 += 64) {
    /* cooperative staging of K/V chunk */
    *(h8*)(ldsK + jrK * 32 + c8K) = *(const h8*)(kSrc + (size_t)(j0 + jrK) * O_QKV);
    *(h8*)(ldsV + dV * 64 + j8V)  = *(const h8*)(vSrc + j0);
    __syncthreads();

    /* S = q K^T over 4 j-tiles */
    v8f z = {};
    v8f s0 = WMMA_F16(a_q, *(const v16h*)(ldsK + (0 * 16 + l16) * 32 + hf * 16), z);
    v8f s1 = WMMA_F16(a_q, *(const v16h*)(ldsK + (1 * 16 + l16) * 32 + hf * 16), z);
    v8f s2 = WMMA_F16(a_q, *(const v16h*)(ldsK + (2 * 16 + l16) * 32 + hf * 16), z);
    v8f s3 = WMMA_F16(a_q, *(const v16h*)(ldsK + (3 * 16 + l16) * 32 + hf * 16), z);

    /* online softmax; C-layout: VGPR r holds row (r + 8*hf), col l16 */
    float nm[8];
#pragma unroll
    for (int r = 0; r < 8; ++r) {
      float t = fmaxf(fmaxf(s0[r], s1[r]), fmaxf(s2[r], s3[r]));
      t = fmaxf(t, __shfl_xor(t, 1, 32));
      t = fmaxf(t, __shfl_xor(t, 2, 32));
      t = fmaxf(t, __shfl_xor(t, 4, 32));
      t = fmaxf(t, __shfl_xor(t, 8, 32));
      nm[r] = fmaxf(m_run[r], t);
      float alpha = __expf(m_run[r] - nm[r]);
      m_run[r] = nm[r];
      l_run[r] *= alpha;
      o0[r] *= alpha;
      o1[r] *= alpha;
    }
    _Float16* pw = ldsP + w * 1024;
#pragma unroll
    for (int r = 0; r < 8; ++r) {
      float p0 = __expf(s0[r] - nm[r]);
      float p1 = __expf(s1[r] - nm[r]);
      float p2 = __expf(s2[r] - nm[r]);
      float p3 = __expf(s3[r] - nm[r]);
      float rs = (p0 + p1) + (p2 + p3);
      rs += __shfl_xor(rs, 1, 32);
      rs += __shfl_xor(rs, 2, 32);
      rs += __shfl_xor(rs, 4, 32);
      rs += __shfl_xor(rs, 8, 32);
      l_run[r] += rs;
      _Float16* pb = pw + (r + hf * 8) * 64 + l16;
      pb[0]  = (_Float16)p0;
      pb[16] = (_Float16)p1;
      pb[32] = (_Float16)p2;
      pb[48] = (_Float16)p3;
    }
    __syncthreads();

    /* O += P V^T : K=64 as two K=32 chunks, N=32 as two d-tiles */
#pragma unroll
    for (int ks = 0; ks < 2; ++ks) {
      const _Float16* pp = pw + l16 * 64 + ks * 32 + hf * 16;
      v16h a_p = cat8(*(const h8*)pp, *(const h8*)(pp + 8));
      v16h bv0 = *(const v16h*)(ldsV + (l16)      * 64 + ks * 32 + hf * 16);
      v16h bv1 = *(const v16h*)(ldsV + (16 + l16) * 64 + ks * 32 + hf * 16);
      o0 = WMMA_F16(a_p, bv0, o0);
      o1 = WMMA_F16(a_p, bv1, o1);
    }
    __syncthreads();   /* protect ldsK/ldsV before restaging */
  }

  /* normalize and store transposed: outT[i][head*32+d] */
#pragma unroll
  for (int r = 0; r < 8; ++r) {
    float inv = 1.0f / l_run[r];
    int i = i0 + r + hf * 8;
    outT[(size_t)i * C_DIM + h * DHEAD + l16]      = (_Float16)(o0[r] * inv);
    outT[(size_t)i * C_DIM + h * DHEAD + 16 + l16] = (_Float16)(o1[r] * inv);
  }
}

/* ---------------- Output projection: y[c][n] = W_out @ out + b ---------- */
__global__ __launch_bounds__(256) void gemm_out(const _Float16* __restrict__ w16,
                                                const _Float16* __restrict__ outT,
                                                const float* __restrict__ bout,
                                                float* __restrict__ y) {
  const int lane = threadIdx.x;
  const int omac = threadIdx.y;                     // 8 c-macros of 32 rows
  const int nmac = blockIdx.x;                      // 128 n-macros of 32 cols
  const int hf  = lane >> 4;
  const int l16 = lane & 15;
  const int arow0 = omac * 32 + l16, arow1 = arow0 + 16;
  const int bcol0 = nmac * 32 + l16, bcol1 = bcol0 + 16;
  v8f acc00 = {}, acc01 = {}, acc10 = {}, acc11 = {};
#pragma unroll
  for (int kk = 0; kk < C_DIM; kk += 32) {
    const h8* ap0 = (const h8*)(w16 + (size_t)arow0 * C_DIM + kk + hf * 16);
    const h8* ap1 = (const h8*)(w16 + (size_t)arow1 * C_DIM + kk + hf * 16);
    v16h a0 = cat8(ap0[0], ap0[1]);
    v16h a1 = cat8(ap1[0], ap1[1]);
    v16h b0 = *(const v16h*)(outT + (size_t)bcol0 * C_DIM + kk + hf * 16);
    v16h b1 = *(const v16h*)(outT + (size_t)bcol1 * C_DIM + kk + hf * 16);
    acc00 = WMMA_F16(a0, b0, acc00);
    acc01 = WMMA_F16(a0, b1, acc01);
    acc10 = WMMA_F16(a1, b0, acc10);
    acc11 = WMMA_F16(a1, b1, acc11);
  }
  const int ob0 = omac * 32 + hf * 8, ob1 = ob0 + 16;
  const int n0 = nmac * 32 + l16, n1 = n0 + 16;
  v8f accs[4] = {acc00, acc01, acc10, acc11};
  const int obs[4] = {ob0, ob0, ob1, ob1};
  const int ns[4]  = {n0, n1, n0, n1};
#pragma unroll
  for (int t = 0; t < 4; ++t) {
#pragma unroll
    for (int i = 0; i < 8; ++i)
      y[(size_t)(obs[t] + i) * N_DIM + ns[t]] = accs[t][i] + bout[obs[t] + i];
  }
}

/* ---------------- launcher --------------------------------------------- */
extern "C" void kernel_launch(void* const* d_in, const int* in_sizes, int n_in,
                              void* d_out, int out_size, void* d_ws, size_t ws_size,
                              hipStream_t stream) {
  const float* x    = (const float*)d_in[0];
  const float* gnw  = (const float*)d_in[1];
  const float* gnb  = (const float*)d_in[2];
  const float* wqkv = (const float*)d_in[3];
  const float* wout = (const float*)d_in[4];
  const float* bout = (const float*)d_in[5];
  float* y = (float*)d_out;

  char* ws = (char*)d_ws;
  float*    sbuf = (float*)(ws + 0);             /* 1 KB  */
  float*    tbuf = (float*)(ws + 1024);          /* 1 KB  */
  _Float16* wq16 = (_Float16*)(ws + 2048);       /* 384 KB */
  _Float16* wo16 = (_Float16*)(ws + 395264);     /* 128 KB */
  _Float16* xnT  = (_Float16*)(ws + 526336);     /* 2 MB  */
  _Float16* qkvT = (_Float16*)(ws + 2623488);    /* 6 MB  */
  _Float16* vN   = (_Float16*)(ws + 8914944);    /* 2 MB  */
  _Float16* outT = (_Float16*)(ws + 11012096);   /* 2 MB  */

  gn_stats<<<8, 256, 0, stream>>>(x, gnw, gnb, sbuf, tbuf);
  cvt_wqkv<<<768, 256, 0, stream>>>(wqkv, wq16);
  cvt_wout<<<256, 256, 0, stream>>>(wout, wo16);
  cvt_x<<<4096, 256, 0, stream>>>(x, sbuf, tbuf, xnT);
  gemm_qkv<<<dim3(128, 3), dim3(32, 8), 0, stream>>>(wq16, xnT, qkvT, vN);
  flash_attn<<<dim3(32, 8), dim3(32, 8), 0, stream>>>(qkvT, vN, outT);
  gemm_out<<<dim3(128, 1), dim3(32, 8), 0, stream>>>(wo16, outT, bout, y);
}
